// LigerFusedLinearCPOLoss_38646115729995
// MI455X (gfx1250) — matrix operand order repaired and checked by reading, not compiled, not run on hardware
//
#include <hip/hip_runtime.h>
#include <hip/hip_bf16.h>

typedef __bf16 bf16;
typedef __attribute__((ext_vector_type(16))) __bf16 v16bf;
typedef __attribute__((ext_vector_type(8)))  __bf16 v8bf;
typedef __attribute__((ext_vector_type(8)))  float   v8f;

// Global-address-space pointer: forces global_load_b128 (LOADcnt only),
// never flat_load_b128 (which would also tick DScnt and entangle with the
// LDS dependency chain that gates the WMMAs).
typedef const __attribute__((address_space(1))) v8bf* gv8bfp;
__device__ __forceinline__ v8bf load_global_v8bf(const bf16* p) {
    return *(gv8bfp)(unsigned long long)(const void*)p;
}

// Problem constants (from reference)
#define BSEQ 8
#define TLEN 1024
#define HDIM 4096
#define VOCAB 32000
#define MTOT (BSEQ * TLEN)          // 8192 token rows
#define BM 128
#define BN 128
#define KT 64                        // K-slab per LDS stage (2 WMMA k-steps)
#define NCH (VOCAB / BN)             // 250 vocab chunks
#define LDK (KT + 16)                // padded LDS stride (elements)
#define NEG_INF (-3.4028235e38f)

__device__ __forceinline__ unsigned short f32_to_bf16_bits(float f) {
    unsigned u = __float_as_uint(f);
    return (unsigned short)((u + 0x7FFFu + ((u >> 16) & 1u)) >> 16);
}

// ---------------- Stage 0: fp32 -> bf16 conversion (vectorized x4) ----------
__global__ void cvt_f32_bf16_kernel(const float* __restrict__ src,
                                    unsigned short* __restrict__ dst, long n4) {
    long i = blockIdx.x * (long)blockDim.x + threadIdx.x;
    long stride = (long)gridDim.x * blockDim.x;
    for (; i < n4; i += stride) {
        float4 f = ((const float4*)src)[i];
        unsigned lo = ((unsigned)f32_to_bf16_bits(f.y) << 16) | f32_to_bf16_bits(f.x);
        unsigned hi = ((unsigned)f32_to_bf16_bits(f.w) << 16) | f32_to_bf16_bits(f.z);
        ((uint2*)dst)[i] = make_uint2(lo, hi);
    }
}

// ---------------- Stage 1: WMMA GEMM + fused per-chunk logsumexp partials ---
__global__ void __launch_bounds__(256)
gemm_lse_kernel(const bf16* __restrict__ X,      // [MTOT][HDIM] bf16
                const bf16* __restrict__ W,      // [VOCAB][HDIM] bf16
                const float* __restrict__ bias,  // [VOCAB]
                const int*  __restrict__ target, // [MTOT]
                float* __restrict__ pmax,        // [NCH][MTOT] chunk-major
                float* __restrict__ psum,        // [NCH][MTOT]
                float* __restrict__ tlogit)      // [MTOT]
{
    __shared__ bf16 Alds[BM * LDK];
    __shared__ bf16 Blds[BN * LDK];
    __shared__ float redmax[2 * BM];
    __shared__ float redsum[2 * BM];

    const int m0   = blockIdx.x * BM;
    const int n0   = blockIdx.y * BN;
    const int tid  = threadIdx.x;
    const int lane = tid & 31;      // wave32 lane
    const int wave = tid >> 5;      // 0..7
    const int wm   = wave >> 1;     // 0..3 (M direction)
    const int wn   = wave & 1;      // 0..1 (N direction)
    const int lrow = lane & 15;
    const int lhi  = lane >> 4;     // 0/1 half-wave

    // ---- per-thread staging geometry: 4 (row,kc) groups of 8 bf16 each ----
    long baseA[4], baseB[4];
    int  ldsOff[4];
#pragma unroll
    for (int g = 0; g < 4; ++g) {
        int gi  = tid + g * 256;
        int row = gi >> 3;           // 0..127
        int kc  = (gi & 7) * 8;      // 0..56
        baseA[g] = (long)(m0 + row) * HDIM + kc;
        baseB[g] = (long)(n0 + row) * HDIM + kc;
        ldsOff[g] = row * LDK + kc;
    }

    v8f acc[2][4];
    const v8f z8 = {0.f,0.f,0.f,0.f,0.f,0.f,0.f,0.f};
#pragma unroll
    for (int i = 0; i < 2; ++i)
#pragma unroll
        for (int j = 0; j < 4; ++j) acc[i][j] = z8;

    // prefetch registers for the next slab (register double buffer)
    v8bf pfA[4], pfB[4];
#pragma unroll
    for (int g = 0; g < 4; ++g) {
        pfA[g] = load_global_v8bf(&X[baseA[g]]);
        pfB[g] = load_global_v8bf(&W[baseB[g]]);
    }

    for (int k0 = 0; k0 < HDIM; k0 += KT) {
        __syncthreads();             // previous compute done -> LDS reusable
#pragma unroll
        for (int g = 0; g < 4; ++g) {
            *(v8bf*)&Alds[ldsOff[g]] = pfA[g];
            *(v8bf*)&Blds[ldsOff[g]] = pfB[g];
        }
        __syncthreads();

        // Issue next slab's global loads now; their s_wait_loadcnt lands at
        // the next iteration's ds_store, so VMEM latency overlaps the WMMAs.
        if (k0 + KT < HDIM) {
            int kn = k0 + KT;
#pragma unroll
            for (int g = 0; g < 4; ++g) {
                pfA[g] = load_global_v8bf(&X[baseA[g] + kn]);
                pfB[g] = load_global_v8bf(&W[baseB[g] + kn]);
            }
        }

#pragma unroll
        for (int kk = 0; kk < KT; kk += 32) {
            v16bf afrag[2];
            v16bf bfrag[4];
            // A 16x32 bf16: lane half picks K base (0/8); VGPR0-3 = K base..base+7,
            // VGPR4-7 = K base+16..base+23 -> two 16B LDS loads.
#pragma unroll
            for (int mi = 0; mi < 2; ++mi) {
                const bf16* p = &Alds[(wm * 32 + mi * 16 + lrow) * LDK + kk + lhi * 8];
                v8bf lo = *(const v8bf*)p;
                v8bf hi = *(const v8bf*)(p + 16);
                afrag[mi] = __builtin_shufflevector(lo, hi, 0,1,2,3,4,5,6,7,8,9,10,11,12,13,14,15);
            }
            // B 32x16 bf16: lane = column; same K order as A (base kk+lhi*8: K base..base+7, base+16..base+23)
#pragma unroll
            for (int ni = 0; ni < 4; ++ni) {
                const bf16* p = &Blds[(wn * 64 + ni * 16 + lrow) * LDK + kk + lhi * 8];
                v8bf lo = *(const v8bf*)p;
                v8bf hi = *(const v8bf*)(p + 16);
                bfrag[ni] = __builtin_shufflevector(lo, hi, 0,1,2,3,4,5,6,7,8,9,10,11,12,13,14,15);
            }
#pragma unroll
            for (int mi = 0; mi < 2; ++mi)
#pragma unroll
                for (int ni = 0; ni < 4; ++ni)
                    acc[mi][ni] = __builtin_amdgcn_wmma_f32_16x16x32_bf16(
                        false, afrag[mi], false, bfrag[ni],
                        (short)0, acc[mi][ni], false, false);
        }
    }

    // ---- epilogue: bias add ----
#pragma unroll
    for (int ni = 0; ni < 4; ++ni) {
        float bv = bias[n0 + wn * 64 + ni * 16 + lrow];
#pragma unroll
        for (int mi = 0; mi < 2; ++mi)
#pragma unroll
            for (int r = 0; r < 8; ++r) acc[mi][ni][r] += bv;
    }

    // ---- target-logit capture (exactly one lane in the grid matches) ----
#pragma unroll
    for (int mi = 0; mi < 2; ++mi)
#pragma unroll
        for (int r = 0; r < 8; ++r) {
            int mrow = m0 + wm * 32 + mi * 16 + r + lhi * 8;
            int tg = target[mrow];
#pragma unroll
            for (int ni = 0; ni < 4; ++ni) {
                int ncol = n0 + wn * 64 + ni * 16 + lrow;
                if (tg == ncol) tlogit[mrow] = acc[mi][ni][r];
            }
        }

    // ---- per-row max & sum(exp) over this wave's 64 columns ----
    // C layout: lanes 0-15 hold M=r, lanes 16-31 hold M=r+8; xor 1..8 stays in-half.
#pragma unroll
    for (int mi = 0; mi < 2; ++mi)
#pragma unroll
        for (int r = 0; r < 8; ++r) {
            float mval = NEG_INF;
#pragma unroll
            for (int ni = 0; ni < 4; ++ni) mval = fmaxf(mval, acc[mi][ni][r]);
#pragma unroll
            for (int off = 1; off < 16; off <<= 1)
                mval = fmaxf(mval, __shfl_xor(mval, off, 32));
            float s = 0.f;
#pragma unroll
            for (int ni = 0; ni < 4; ++ni) s += __expf(acc[mi][ni][r] - mval);
#pragma unroll
            for (int off = 1; off < 16; off <<= 1)
                s += __shfl_xor(s, off, 32);
            if (lrow == 0) {
                int ml = wm * 32 + mi * 16 + r + lhi * 8;
                redmax[wn * BM + ml] = mval;
                redsum[wn * BM + ml] = s;
            }
        }
    __syncthreads();

    if (tid < BM) {
        float m1 = redmax[tid],      m2 = redmax[BM + tid];
        float s1 = redsum[tid],      s2 = redsum[BM + tid];
        float Mx = fmaxf(m1, m2);
        float S  = s1 * __expf(m1 - Mx) + s2 * __expf(m2 - Mx);
        pmax[(long)blockIdx.y * MTOT + m0 + tid] = Mx;
        psum[(long)blockIdx.y * MTOT + m0 + tid] = S;
    }
}

// ---------------- Stage 2: combine chunk partials -> per-token logprob ------
__global__ void lse_kernel(const float* __restrict__ pmax,
                           const float* __restrict__ psum,
                           const float* __restrict__ tlogit,
                           const int*  __restrict__ target,
                           float* __restrict__ tok_logp) {
    int m = blockIdx.x * blockDim.x + threadIdx.x;
    if (m >= MTOT) return;
    float Mx = NEG_INF, S = 0.f;
    for (int c = 0; c < NCH; ++c) {
        float mc = pmax[(long)c * MTOT + m];
        float sc = psum[(long)c * MTOT + m];
        float Mn = fmaxf(Mx, mc);
        S = S * __expf(Mx - Mn) + sc * __expf(mc - Mn);
        Mx = Mn;
    }
    float lse = Mx + logf(S);
    bool msk = (target[m] != -100);
    tok_logp[m] = msk ? (tlogit[m] - lse) : 0.f;
}

// ---------------- Stage 3: sequence averages + CPO scalar loss --------------
__device__ __forceinline__ float block_reduce_sum_1024(float v, float* tmp) {
#pragma unroll
    for (int off = 16; off > 0; off >>= 1) v += __shfl_xor(v, off, 32);
    int lane = threadIdx.x & 31, w = threadIdx.x >> 5;
    if (lane == 0) tmp[w] = v;
    __syncthreads();
    float r = 0.f;
    if (w == 0) {
        r = tmp[lane];
#pragma unroll
        for (int off = 16; off > 0; off >>= 1) r += __shfl_xor(r, off, 32);
    }
    __syncthreads();
    return r;  // valid in wave 0
}

__global__ void __launch_bounds__(1024)
final_loss_kernel(const float* __restrict__ tok_logp,
                  const int*  __restrict__ target,
                  float* __restrict__ out) {
    __shared__ float tmp[32];
    __shared__ float res[16];
    int t = threadIdx.x;  // token within sequence (TLEN == 1024)
    for (int b = 0; b < BSEQ; ++b) {
        float v = tok_logp[b * TLEN + t];
        float c = (target[b * TLEN + t] != -100) ? 1.f : 0.f;
        float sv = block_reduce_sum_1024(v, tmp);
        float sc = block_reduce_sum_1024(c, tmp);
        if (t == 0) { res[b] = sv; res[8 + b] = sc; }
    }
    __syncthreads();
    if (t == 0) {
        const float BETA = 0.1f, ALPHA = 1.0f;
        float avg[BSEQ];
        for (int b = 0; b < BSEQ; ++b) avg[b] = res[b] / res[8 + b];
        float pref = 0.f;
        for (int i = 0; i < 4; ++i) {
            float pl = BETA * (avg[i] - avg[i + 4]);
            // -log_sigmoid(pl) == softplus(-pl), numerically stable
            float zv = -pl;
            pref += fmaxf(zv, 0.f) + log1pf(expf(-fabsf(zv)));
        }
        pref /= 4.f;
        float nll_num = -(res[0] + res[1] + res[2] + res[3]);
        float nll_den =  (res[8] + res[9] + res[10] + res[11]);
        out[0] = ALPHA * (nll_num / nll_den) + pref;
    }
}

// ---------------------------------------------------------------------------
extern "C" void kernel_launch(void* const* d_in, const int* in_sizes, int n_in,
                              void* d_out, int out_size, void* d_ws, size_t ws_size,
                              hipStream_t stream) {
    const float* lin_weight = (const float*)d_in[0];  // [VOCAB][HDIM]
    const float* x          = (const float*)d_in[1];  // [BSEQ][TLEN][HDIM]
    const int*   target     = (const int*)d_in[2];    // [BSEQ][TLEN]
    const float* bias       = (const float*)d_in[3];  // [VOCAB]
    float* out = (float*)d_out;

    char* ws = (char*)d_ws;
    size_t off = 0;
    bf16* Xbf = (bf16*)(ws + off);            off += (size_t)MTOT  * HDIM * 2;
    bf16* Wbf = (bf16*)(ws + off);            off += (size_t)VOCAB * HDIM * 2;
    float* pmax   = (float*)(ws + off);       off += (size_t)NCH * MTOT * 4;
    float* psum   = (float*)(ws + off);       off += (size_t)NCH * MTOT * 4;
    float* tlogit = (float*)(ws + off);       off += (size_t)MTOT * 4;
    float* tokLp  = (float*)(ws + off);       off += (size_t)MTOT * 4;

    // Stage 0: fp32 -> bf16 (X then W)
    cvt_f32_bf16_kernel<<<4096, 256, 0, stream>>>(x, (unsigned short*)Xbf,
                                                  (long)MTOT * HDIM / 4);
    cvt_f32_bf16_kernel<<<16384, 256, 0, stream>>>(lin_weight, (unsigned short*)Wbf,
                                                   (long)VOCAB * HDIM / 4);
    // Stage 1: fused GEMM + chunk logsumexp partials.
    // grid.x = M-blocks fastest so all M-blocks of one vocab chunk share the
    // weight tile via L2; X (67 MB bf16) stays L2-resident across chunks.
    dim3 grid(MTOT / BM, NCH);
    gemm_lse_kernel<<<grid, 256, 0, stream>>>(Xbf, Wbf, bias, target,
                                              pmax, psum, tlogit);
    // Stage 2: per-token logprobs
    lse_kernel<<<MTOT / 256, 256, 0, stream>>>(pmax, psum, tlogit, target, tokLp);
    // Stage 3: scalar CPO loss
    final_loss_kernel<<<1, 1024, 0, stream>>>(tokLp, target, out);
}
